// RGCNPolypharmacy_24180665876651
// MI455X (gfx1250) — matrix operand order repaired and checked by reading, not compiled, not run on hardware
//
#include <hip/hip_runtime.h>
#include <hip/hip_bf16.h>

#define HDIM 256

typedef _Float16 half16 __attribute__((ext_vector_type(16)));
typedef _Float16 half8  __attribute__((ext_vector_type(8)));
typedef _Float16 half4  __attribute__((ext_vector_type(4)));
typedef float    float8 __attribute__((ext_vector_type(8)));

// ---------------------------------------------------------------------------
// f32 -> f16 conversion (4 elements per thread)
// ---------------------------------------------------------------------------
__global__ void cvt_f16_kernel(const float* __restrict__ x,
                               _Float16* __restrict__ xh, int n4) {
  int i = blockIdx.x * blockDim.x + threadIdx.x;
  if (i < n4) {
    float4 v = ((const float4*)x)[i];
    half4 o;
    o[0] = (_Float16)v.x; o[1] = (_Float16)v.y;
    o[2] = (_Float16)v.z; o[3] = (_Float16)v.w;
    ((half4*)xh)[i] = o;
  }
}

// ---------------------------------------------------------------------------
// Build 9 concatenated f16 weight matrices:
//   m=0..5 : W_r = sum_b softmax(coeff[r])[b] * basis[b]   (H x H, [out][in])
//   m=6..8 : Wself_drug / Wself_protein / Wself_se
// grid.x = 9*256 blocks of 256 threads (one element per thread)
// ---------------------------------------------------------------------------
__global__ void make_weights_kernel(const float* __restrict__ basis,
                                    const float* __restrict__ coeff,
                                    const float* __restrict__ Wd,
                                    const float* __restrict__ Wp,
                                    const float* __restrict__ Ws,
                                    _Float16* __restrict__ Wcat) {
  const int HH = HDIM * HDIM;
  int m   = blockIdx.x >> 8;
  int idx = ((blockIdx.x & 255) << 8) + threadIdx.x;   // 0..65535
  float v;
  if (m < 6) {
    float c0 = coeff[m * 4 + 0], c1 = coeff[m * 4 + 1];
    float c2 = coeff[m * 4 + 2], c3 = coeff[m * 4 + 3];
    float mx = fmaxf(fmaxf(c0, c1), fmaxf(c2, c3));
    float e0 = expf(c0 - mx), e1 = expf(c1 - mx);
    float e2 = expf(c2 - mx), e3 = expf(c3 - mx);
    float inv = 1.0f / (e0 + e1 + e2 + e3);
    v = (e0 * basis[idx] + e1 * basis[HH + idx] +
         e2 * basis[2 * HH + idx] + e3 * basis[3 * HH + idx]) * inv;
  } else if (m == 6) v = Wd[idx];
  else if (m == 7)   v = Wp[idx];
  else               v = Ws[idx];
  Wcat[(size_t)m * HH + idx] = (_Float16)v;
}

// ---------------------------------------------------------------------------
// WMMA fragment helpers (wave32; contiguous k order, same for A and B)
// A (16x32, MxK): lane l -> row M=l%16; halves [0..15] = K + (l<16?0:16) ..
// B (32x16, KxN): lane l -> col N=l%16; halves [0..15] = K + (l<16?0:16) ..
// ---------------------------------------------------------------------------
__device__ __forceinline__ half16 load_a_frag(const _Float16* __restrict__ X,
                                              int row, int k, int hi) {
  return *(const half16*)(X + (size_t)row * HDIM + k + hi * 16);
}

__device__ __forceinline__ half16 load_b_frag(const _Float16* __restrict__ W,
                                              int ncol, int k, int hi) {
  return *(const half16*)(W + (size_t)ncol * HDIM + k + hi * 16);
}

// ---------------------------------------------------------------------------
// OUT[m, n] = sum_k X[m,k] * W[n,k]    (X: n_rows x 256 f16, W: 256x256 f16
//  stored [out][in], OUT: n_rows x 256 f32)
// block = 256 threads (8 waves); wave w owns N-tiles 2w, 2w+1; grid.x = rows/16
// ---------------------------------------------------------------------------
__global__ void gemm_kernel(const _Float16* __restrict__ X,
                            const _Float16* __restrict__ W,
                            float* __restrict__ OUT, int n_rows) {
  int wave = threadIdx.x >> 5;
  int lane = threadIdx.x & 31;
  int l15  = lane & 15;
  int hi   = lane >> 4;
  int mbase = blockIdx.x * 16;
  int mrow  = mbase + l15;
  if (mrow >= n_rows) mrow = n_rows - 1;   // branchless-ish clamp (EXEC stays full)
  int n0 = wave * 32;
  int n1 = n0 + 16;

  float8 c0 = {};
  float8 c1 = {};
#pragma unroll
  for (int k = 0; k < HDIM; k += 32) {
    half16 a  = load_a_frag(X, mrow, k, hi);
    half16 b0 = load_b_frag(W, n0 + l15, k, hi);
    half16 b1 = load_b_frag(W, n1 + l15, k, hi);
    c0 = __builtin_amdgcn_wmma_f32_16x16x32_f16(false, a, false, b0, (short)0,
                                                c0, false, false);
    c1 = __builtin_amdgcn_wmma_f32_16x16x32_f16(false, a, false, b1, (short)0,
                                                c1, false, false);
  }
  // D layout: VGPR v -> row M = v + (l<16?0:8), col N = l%16
#pragma unroll
  for (int v = 0; v < 8; ++v) {
    int m = mbase + v + hi * 8;
    if (m < n_rows) {
      OUT[(size_t)m * HDIM + n0 + l15] = c0[v];
      OUT[(size_t)m * HDIM + n1 + l15] = c1[v];
    }
  }
}

// ---------------------------------------------------------------------------
// Edge scatter: agg[dst] += msg[src];  cnt[dst] += 1
// 4 edges per 256-thread block; 64 lanes x float4 per edge.
// msg (<=51MB) + agg (<=51MB) are L2-resident -> atomics resolve in L2.
// ---------------------------------------------------------------------------
__global__ void scatter_kernel(const float* __restrict__ MSG,
                               const int* __restrict__ SRC,
                               const int* __restrict__ DST,
                               float* __restrict__ AGG,
                               float* __restrict__ CNT, int n_edges) {
  int t   = threadIdx.x;
  int sub = t >> 6;           // edge slot within block
  int l   = t & 63;           // 64 lanes * 4 floats = 256 cols
  int e   = blockIdx.x * 4 + sub;
  if (e >= n_edges) return;
  int s = SRC[e];
  int d = DST[e];
  float4 v = ((const float4*)(MSG + (size_t)s * HDIM))[l];
  float* a = AGG + (size_t)d * HDIM + l * 4;
  atomicAdd(a + 0, v.x);
  atomicAdd(a + 1, v.y);
  atomicAdd(a + 2, v.z);
  atomicAdd(a + 3, v.w);
  if (l == 0) atomicAdd(CNT + d, 1.0f);
}

// ---------------------------------------------------------------------------
// Fused epilogue: h_self = X @ Wself.T (WMMA), h = relu(h_self + agg/deg) + x,
// out = LayerNorm(h) * g + b.   One block = 16 rows x 256 cols.
// ---------------------------------------------------------------------------
__global__ void combine_kernel(const float* __restrict__ X,
                               const _Float16* __restrict__ Xh,
                               const _Float16* __restrict__ Wself,
                               const float* __restrict__ AGG,
                               const float* __restrict__ CNT,
                               const float* __restrict__ G,
                               const float* __restrict__ B,
                               float* __restrict__ OUT, int n_rows) {
  __shared__ float sm[16 * HDIM];
  int wave = threadIdx.x >> 5;
  int lane = threadIdx.x & 31;
  int l15  = lane & 15;
  int hi   = lane >> 4;
  int mbase = blockIdx.x * 16;
  int mrow  = mbase + l15;
  if (mrow >= n_rows) mrow = n_rows - 1;
  int n0 = wave * 32;
  int n1 = n0 + 16;

  float8 c0 = {};
  float8 c1 = {};
#pragma unroll
  for (int k = 0; k < HDIM; k += 32) {
    half16 a  = load_a_frag(Xh, mrow, k, hi);
    half16 b0 = load_b_frag(Wself, n0 + l15, k, hi);
    half16 b1 = load_b_frag(Wself, n1 + l15, k, hi);
    c0 = __builtin_amdgcn_wmma_f32_16x16x32_f16(false, a, false, b0, (short)0,
                                                c0, false, false);
    c1 = __builtin_amdgcn_wmma_f32_16x16x32_f16(false, a, false, b1, (short)0,
                                                c1, false, false);
  }
#pragma unroll
  for (int v = 0; v < 8; ++v) {
    int mloc = v + hi * 8;
    int m = mbase + mloc;
    int mc = (m < n_rows) ? m : (n_rows - 1);
    float invdeg = 1.0f / fmaxf(CNT[mc], 1.0f);
    int colA = n0 + l15, colB = n1 + l15;
    float hA = fmaxf(c0[v] + AGG[(size_t)mc * HDIM + colA] * invdeg, 0.0f) +
               X[(size_t)mc * HDIM + colA];
    float hB = fmaxf(c1[v] + AGG[(size_t)mc * HDIM + colB] * invdeg, 0.0f) +
               X[(size_t)mc * HDIM + colB];
    sm[mloc * HDIM + colA] = hA;
    sm[mloc * HDIM + colB] = hB;
  }
  __syncthreads();

  // LayerNorm: wave w handles rows 2w and 2w+1; 8 elements per lane.
#pragma unroll
  for (int rr = 0; rr < 2; ++rr) {
    int r  = wave * 2 + rr;
    int gm = mbase + r;
    float vals[8];
    float s = 0.0f, ss = 0.0f;
#pragma unroll
    for (int i = 0; i < 8; ++i) {
      float v = sm[r * HDIM + lane * 8 + i];
      vals[i] = v;
      s += v;
      ss += v * v;
    }
#pragma unroll
    for (int off = 16; off > 0; off >>= 1) {
      s  += __shfl_xor(s, off, 32);
      ss += __shfl_xor(ss, off, 32);
    }
    float mean = s * (1.0f / HDIM);
    float var  = ss * (1.0f / HDIM) - mean * mean;
    float rs   = rsqrtf(var + 1e-5f);
    if (gm < n_rows) {
#pragma unroll
      for (int i = 0; i < 8; ++i) {
        int col = lane * 8 + i;
        OUT[(size_t)gm * HDIM + col] = (vals[i] - mean) * rs * G[col] + B[col];
      }
    }
  }
}

// ---------------------------------------------------------------------------
// Host orchestration
// ---------------------------------------------------------------------------
extern "C" void kernel_launch(void* const* d_in, const int* in_sizes, int n_in,
                              void* d_out, int out_size, void* d_ws, size_t ws_size,
                              hipStream_t stream) {
  (void)n_in; (void)out_size; (void)ws_size;
  const int H = HDIM;
  const float* x[3]   = {(const float*)d_in[0], (const float*)d_in[1],
                         (const float*)d_in[2]};
  const float* basis  = (const float*)d_in[3];
  const float* coeff  = (const float*)d_in[4];
  const float* Wsf[3] = {(const float*)d_in[5], (const float*)d_in[6],
                         (const float*)d_in[7]};
  const float* g[3]   = {(const float*)d_in[8], (const float*)d_in[10],
                         (const float*)d_in[12]};
  const float* b[3]   = {(const float*)d_in[9], (const float*)d_in[11],
                         (const float*)d_in[13]};
  const int* edges[6];
  for (int r = 0; r < 6; ++r) edges[r] = (const int*)d_in[14 + r];
  const int E = in_sizes[14] / 2;

  int N[3] = {in_sizes[0] / H, in_sizes[1] / H, in_sizes[2] / H};
  int Nmax = N[0] > N[1] ? (N[0] > N[2] ? N[0] : N[2]) : (N[1] > N[2] ? N[1] : N[2]);

  // ---- workspace layout (aggs + cnts first so one memset zeroes them) ----
  char* ws = (char*)d_ws;
  size_t off = 0;
  auto take = [&](size_t bytes) -> char* {
    char* p = ws + off;
    off += (bytes + 255) & ~(size_t)255;
    return p;
  };
  float* agg[3]; float* cnt[3];
  for (int t = 0; t < 3; ++t) agg[t] = (float*)take((size_t)N[t] * H * 4);
  for (int t = 0; t < 3; ++t) cnt[t] = (float*)take((size_t)N[t] * 4);
  size_t zero_bytes = off;
  _Float16* xh[3];
  for (int t = 0; t < 3; ++t) xh[t] = (_Float16*)take((size_t)N[t] * H * 2);
  _Float16* Wcat = (_Float16*)take((size_t)9 * H * H * 2);
  float* msgbuf  = (float*)take((size_t)Nmax * H * 4);

  hipMemsetAsync(d_ws, 0, zero_bytes, stream);

  // f32 -> f16 node features
  for (int t = 0; t < 3; ++t) {
    int n4 = N[t] * H / 4;
    cvt_f16_kernel<<<(n4 + 255) / 256, 256, 0, stream>>>(x[t], xh[t], n4);
  }
  // 6 relation matrices + 3 self matrices in f16
  make_weights_kernel<<<9 * 256, 256, 0, stream>>>(basis, coeff, Wsf[0], Wsf[1],
                                                   Wsf[2], Wcat);

  // relation r: src type / dst type
  const int src_of[6] = {0, 0, 1, 1, 0, 2};
  const int dst_of[6] = {0, 1, 0, 1, 2, 0};
  for (int r = 0; r < 6; ++r) {
    int st = src_of[r], dt = dst_of[r];
    gemm_kernel<<<(N[st] + 15) / 16, 256, 0, stream>>>(
        xh[st], Wcat + (size_t)r * H * H, msgbuf, N[st]);
    scatter_kernel<<<(E + 3) / 4, 256, 0, stream>>>(
        msgbuf, edges[r], edges[r] + E, agg[dt], cnt[dt], E);
  }

  // fused self-GEMM + relu + residual + LayerNorm
  float* outp = (float*)d_out;
  for (int t = 0; t < 3; ++t) {
    combine_kernel<<<(N[t] + 15) / 16, 256, 0, stream>>>(
        x[t], xh[t], Wcat + (size_t)(6 + t) * H * H, agg[t], cnt[t], g[t], b[t],
        outp, N[t]);
    outp += (size_t)N[t] * H;
  }
}
